// TemEnc_56461640073568
// MI455X (gfx1250) — matrix operand order repaired and checked by reading, not compiled, not run on hardware
//
#include <hip/hip_runtime.h>
#include <hip/hip_bf16.h>
#include <math.h>

// ---------------------------------------------------------------------------
// MI455X (gfx1250, wave32) masked-autoencoder transformer.
// All GEMMs + attention on v_wmma_f32_16x16x32_f16; packed b128 LDS staging;
// global_prefetch_b8 for next-tile prefetch.
// ---------------------------------------------------------------------------

typedef _Float16 v8h  __attribute__((ext_vector_type(8)));
typedef _Float16 v16h __attribute__((ext_vector_type(16)));
typedef float    v8f  __attribute__((ext_vector_type(8)));

#define BB 8
#define LL 1024
#define CC 55
#define DD 256
#define FF 1024
#define HH 8
#define TR 512
#define WIN 12
#define NEG_LOG1E4_OVER_D (-9.210340371976184f / 256.0f)

__device__ inline v8h cvt8(float4 a, float4 b) {
  v8h r;
  r[0] = (_Float16)a.x; r[1] = (_Float16)a.y;
  r[2] = (_Float16)a.z; r[3] = (_Float16)a.w;
  r[4] = (_Float16)b.x; r[5] = (_Float16)b.y;
  r[6] = (_Float16)b.z; r[7] = (_Float16)b.w;
  return r;
}

// ===========================================================================
// 1. Embedding: circular conv1d (k=3, no bias) + sinusoidal PE
// ===========================================================================
__global__ __launch_bounds__(256) void embed_kernel(
    const float* __restrict__ x, const float* __restrict__ cw,
    float* __restrict__ ex)
{
  __shared__ float sx[3 * 64];
  int bl = blockIdx.x;
  int b = bl / LL, l = bl % LL;
  int lm = (l == 0) ? (LL - 1) : (l - 1);
  int lp = (l == LL - 1) ? 0 : (l + 1);
  int tid = threadIdx.x;
  if (tid < CC) {
    sx[tid]       = x[((size_t)(b * LL + lm)) * CC + tid];
    sx[64 + tid]  = x[((size_t)(b * LL + l )) * CC + tid];
    sx[128 + tid] = x[((size_t)(b * LL + lp)) * CC + tid];
  }
  __syncthreads();
  int d = tid;
  float acc = 0.f;
  for (int c = 0; c < CC; ++c) {
    acc += sx[c]       * cw[(0 * CC + c) * DD + d];
    acc += sx[64 + c]  * cw[(1 * CC + c) * DD + d];
    acc += sx[128 + c] * cw[(2 * CC + c) * DD + d];
  }
  int j = d >> 1;
  float freq = __expf((float)(2 * j) * NEG_LOG1E4_OVER_D);
  float ang = (float)l * freq;
  acc += (d & 1) ? __cosf(ang) : __sinf(ang);
  ex[((size_t)(b * LL + l)) * DD + d] = acc;
}

// ===========================================================================
// 2. Causal box-filter mean/var score.  One wave per (b,l).
// ===========================================================================
__global__ __launch_bounds__(128) void score_kernel(
    const float* __restrict__ ex, float* __restrict__ score)
{
  int idx = blockIdx.x * 4 + (threadIdx.x >> 5);  // b*LL + l
  int lane = threadIdx.x & 31;
  int l = idx % LL;
  int w0 = l - (WIN - 1); if (w0 < 0) w0 = 0;
  float cnt = (float)(l - w0 + 1);
  float sv = 0.f, sm = 0.f;
  for (int j = 0; j < 8; ++j) {
    int d = lane + j * 32;
    float s1 = 0.f, s2 = 0.f;
    for (int w = w0; w <= l; ++w) {
      float t = ex[((size_t)(idx - l + w)) * DD + d];
      s1 += t; s2 += t * t;
    }
    float m = s1 / cnt;
    sm += m;
    sv += s2 / cnt - m * m;
  }
  for (int m = 16; m; m >>= 1) {
    sv += __shfl_xor(sv, m, 32);
    sm += __shfl_xor(sm, m, 32);
  }
  if (lane == 0) score[idx] = sv / sm;
}

// ===========================================================================
// 3. Partition by rank (top-k equivalent; jax tie-break: lower index first).
// ===========================================================================
__global__ __launch_bounds__(1024) void rank_kernel(
    const float* __restrict__ score, int* __restrict__ midx,
    int* __restrict__ uidx)
{
  __shared__ float ss[LL];
  int b = blockIdx.x, t = threadIdx.x;
  float my = score[(size_t)b * LL + t];
  ss[t] = my;
  __syncthreads();
  int rank = 0;
  for (int j = 0; j < LL; ++j) {
    float o = ss[j];
    rank += (o > my) || (o == my && j < t);
  }
  if (rank < TR) midx[b * TR + rank] = t;
  else           uidx[b * TR + (LL - 1 - rank)] = t;
}

// ===========================================================================
// 4. Gather / scatter
// ===========================================================================
__global__ __launch_bounds__(256) void gather_kernel(
    const float* __restrict__ ex, const int* __restrict__ uidx,
    float* __restrict__ eu)
{
  int bi = blockIdx.x;
  int b = bi / TR;
  int l = uidx[bi];
  eu[(size_t)bi * DD + threadIdx.x] =
      ex[((size_t)(b * LL + l)) * DD + threadIdx.x];
}

__global__ __launch_bounds__(256) void scatter_u_kernel(
    const float* __restrict__ ux, const int* __restrict__ uidx,
    float* __restrict__ tok)
{
  int bi = blockIdx.x;
  int b = bi / TR;
  int l = uidx[bi];
  tok[((size_t)(b * LL + l)) * DD + threadIdx.x] =
      ux[(size_t)bi * DD + threadIdx.x];
}

__global__ __launch_bounds__(256) void scatter_m_kernel(
    const float* __restrict__ mask_token, const int* __restrict__ midx,
    float* __restrict__ tok)
{
  int bi = blockIdx.x;
  int b = bi / TR;
  int l = midx[bi];
  int d = threadIdx.x;
  int j = d >> 1;
  float freq = __expf((float)(2 * j) * NEG_LOG1E4_OVER_D);
  float ang = (float)l * freq;
  float pe = (d & 1) ? __cosf(ang) : __sinf(ang);
  tok[((size_t)(b * LL + l)) * DD + d] = mask_token[d] + pe;
}

// ===========================================================================
// 5. WMMA GEMM: C[N,M] = act(A[N,K] * W[K,M] + bias)
//    128 threads (4 waves), 128x64 block tile; each wave owns 32 rows
//    (2 A-fragments, 8 f32 accumulators).  f16 inputs, f32 accumulate.
//    act: 0=none 1=relu 2=gelu(exact) 3=sigmoid
// ===========================================================================
__global__ __launch_bounds__(128) void gemm_kernel(
    const float* __restrict__ A, const float* __restrict__ W,
    const float* __restrict__ bias, float* __restrict__ C,
    int N, int K, int M, int act)
{
  __shared__ __align__(16) _Float16 sA[128 * 40];  // [row][k]
  __shared__ __align__(16) _Float16 sB[64 * 40];   // [m][k]  (W transposed)
  int tid = threadIdx.x;
  int wave = tid >> 5, lane = tid & 31;
  int half = lane >> 4, l16 = lane & 15;
  int m0 = blockIdx.x * 64;
  int n0 = blockIdx.y * 128;

  // B-stage ownership: each thread owns one m-column half (16 k values)
  int bcol = tid & 63, bkg = tid >> 6;  // bkg in {0,1}: k 0-15 / 16-31

  v8f acc[2][4] = {};

  for (int kc = 0; kc < K; kc += 32) {
    // ---- prefetch next K tile into L2 (global_prefetch_b8) ----
    if (kc + 32 < K) {
      __builtin_prefetch(A + (size_t)(n0 + tid) * K + kc + 32, 0, 2);
      __builtin_prefetch(W + (size_t)(kc + 32 + bkg * 16) * M + m0 + bcol, 0, 2);
    }
    // ---- stage A tile 128x32 -> f16, one row per thread, 4x b128 stores ----
    {
      const float4* src = (const float4*)(A + (size_t)(n0 + tid) * K + kc);
      float4 f0 = src[0], f1 = src[1], f2 = src[2], f3 = src[3];
      float4 f4 = src[4], f5 = src[5], f6 = src[6], f7 = src[7];
      v8h* dst = (v8h*)(sA + tid * 40);
      dst[0] = cvt8(f0, f1);
      dst[1] = cvt8(f2, f3);
      dst[2] = cvt8(f4, f5);
      dst[3] = cvt8(f6, f7);
    }
    // ---- stage W tile 32x64 transposed: thread owns (col, 16 k's) ----
    {
      float wv[16];
#pragma unroll
      for (int j = 0; j < 16; ++j)
        wv[j] = W[(size_t)(kc + bkg * 16 + j) * M + m0 + bcol];
      v8h lo, hi;
#pragma unroll
      for (int i = 0; i < 8; ++i) { lo[i] = (_Float16)wv[i]; hi[i] = (_Float16)wv[i + 8]; }
      v8h* bdst = (v8h*)(sB + bcol * 40 + bkg * 16);
      bdst[0] = lo;
      bdst[1] = hi;
    }
    __syncthreads();

    // A fragments (16x32): lane<16 K0-7,K16-23; lane>=16 K8-15,K24-31
    v16h afrag[2];
#pragma unroll
    for (int s = 0; s < 2; ++s) {
      int arow = wave * 32 + s * 16 + l16;
      const v8h* ap = (const v8h*)(sA + arow * 40 + half * 8);
      v8h alo = ap[0], ahi = ap[2];
#pragma unroll
      for (int j = 0; j < 8; ++j) { afrag[s][j] = alo[j]; afrag[s][j + 8] = ahi[j]; }
    }

#pragma unroll
    for (int t = 0; t < 4; ++t) {
      int brow = t * 16 + l16;
      const v8h* bp = (const v8h*)(sB + brow * 40 + half * 8);
      v8h b0 = bp[0], b1 = bp[2];
      v16h bfrag;
#pragma unroll
      for (int j = 0; j < 8; ++j) { bfrag[j] = b0[j]; bfrag[j + 8] = b1[j]; }
      acc[0][t] = __builtin_amdgcn_wmma_f32_16x16x32_f16(
          false, afrag[0], false, bfrag, (short)0, acc[0][t], false, false);
      acc[1][t] = __builtin_amdgcn_wmma_f32_16x16x32_f16(
          false, afrag[1], false, bfrag, (short)0, acc[1][t], false, false);
    }
    __syncthreads();
  }

#pragma unroll
  for (int s = 0; s < 2; ++s) {
    int rbase = n0 + wave * 32 + s * 16 + half * 8;
#pragma unroll
    for (int t = 0; t < 4; ++t) {
      int col = m0 + t * 16 + l16;
      float bv = bias ? bias[col] : 0.f;
#pragma unroll
      for (int v = 0; v < 8; ++v) {
        float val = acc[s][t][v] + bv;
        if (act == 1)      val = fmaxf(val, 0.f);
        else if (act == 2) val = 0.5f * val * (1.f + erff(val * 0.7071067811865475f));
        else if (act == 3) val = 1.f / (1.f + __expf(-val));
        C[(size_t)(rbase + v) * M + col] = val;
      }
    }
  }
}

// ===========================================================================
// 6. Flash attention, all-WMMA.  Block = (b,h, 64 query rows), 4 waves.
// ===========================================================================
__global__ __launch_bounds__(128) void attn_kernel(
    const float* __restrict__ Q, const float* __restrict__ K,
    const float* __restrict__ V, float* __restrict__ O,
    int L, float scale)
{
  __shared__ __align__(16) _Float16 sQ[64 * 40];       // [qrow][dh]
  __shared__ __align__(16) _Float16 sK[64 * 40];       // [krow][dh]
  __shared__ __align__(16) _Float16 sVT[32 * 72];      // [dh][krow]
  __shared__ __align__(16) _Float16 sP[4][16 * 72];    // per-wave P tile

  int tid = threadIdx.x;
  int wave = tid >> 5, lane = tid & 31;
  int half = lane >> 4, l16 = lane & 15;
  int bh = blockIdx.y;
  int b = bh / HH, h = bh % HH;
  int q0 = blockIdx.x * 64;

  int sr = tid >> 1, scg = (tid & 1) * 16;  // contiguous-stage ownership
  int vd = tid & 31, vpg = tid >> 5;        // VT-stage ownership (d, pos grp)

  // stage Q tile once (2x b128 stores per thread)
  {
    const float4* src = (const float4*)(Q + ((size_t)(b * L + q0 + sr)) * DD + h * 32 + scg);
    float4 f0 = src[0], f1 = src[1], f2 = src[2], f3 = src[3];
    v8h* dst = (v8h*)(sQ + sr * 40 + scg);
    dst[0] = cvt8(f0, f1);
    dst[1] = cvt8(f2, f3);
  }
  __syncthreads();

  v16h qf;
  {
    int arow = wave * 16 + l16;
    const v8h* ap = (const v8h*)(sQ + arow * 40 + half * 8);
    v8h lo = ap[0], hi = ap[2];
#pragma unroll
    for (int j = 0; j < 8; ++j) { qf[j] = lo[j]; qf[j + 8] = hi[j]; }
  }

  float mrow[8], lrow[8];
  v8f o0 = {}, o1 = {};
#pragma unroll
  for (int v = 0; v < 8; ++v) { mrow[v] = -3.0e38f; lrow[v] = 0.f; }

  int nct = L / 64;
  for (int ct = 0; ct < nct; ++ct) {
    __syncthreads();
    int c0 = ct * 64;
    // prefetch next tile's K/V rows
    if (ct + 1 < nct) {
      __builtin_prefetch(K + ((size_t)(b * L + c0 + 64 + sr)) * DD + h * 32 + scg, 0, 2);
      __builtin_prefetch(V + ((size_t)(b * L + c0 + 64 + sr)) * DD + h * 32 + scg, 0, 2);
    }
    // stage K tile [pos][dh] (contiguous)
    {
      const float4* ksrc = (const float4*)(K + ((size_t)(b * L + c0 + sr)) * DD + h * 32 + scg);
      float4 f0 = ksrc[0], f1 = ksrc[1], f2 = ksrc[2], f3 = ksrc[3];
      v8h* kdst = (v8h*)(sK + sr * 40 + scg);
      kdst[0] = cvt8(f0, f1);
      kdst[1] = cvt8(f2, f3);
    }
    // stage V transposed [dh][pos]: thread owns (dh=vd, 16 positions)
    {
      float vv[16];
#pragma unroll
      for (int j = 0; j < 16; ++j)
        vv[j] = V[((size_t)(b * L + c0 + vpg * 16 + j)) * DD + h * 32 + vd];
      v8h lo, hi;
#pragma unroll
      for (int i = 0; i < 8; ++i) { lo[i] = (_Float16)vv[i]; hi[i] = (_Float16)vv[i + 8]; }
      v8h* vdst = (v8h*)(sVT + vd * 72 + vpg * 16);
      vdst[0] = lo;
      vdst[1] = hi;
    }
    __syncthreads();

    // S = Q * K^T
    v8f s[4];
#pragma unroll
    for (int t = 0; t < 4; ++t) {
      int brow = t * 16 + l16;
      const v8h* bp = (const v8h*)(sK + brow * 40 + half * 8);
      v8h b0 = bp[0], b1 = bp[2];
      v16h bf;
#pragma unroll
      for (int j = 0; j < 8; ++j) { bf[j] = b0[j]; bf[j + 8] = b1[j]; }
      v8f z = {};
      s[t] = __builtin_amdgcn_wmma_f32_16x16x32_f16(
          false, qf, false, bf, (short)0, z, false, false);
    }
#pragma unroll
    for (int t = 0; t < 4; ++t)
#pragma unroll
      for (int v = 0; v < 8; ++v) s[t][v] *= scale;

    // online softmax (row = v + 8*half, replicated across 16 lanes)
#pragma unroll
    for (int v = 0; v < 8; ++v) {
      float tmax = fmaxf(fmaxf(s[0][v], s[1][v]), fmaxf(s[2][v], s[3][v]));
      for (int m = 8; m; m >>= 1) tmax = fmaxf(tmax, __shfl_xor(tmax, m, 32));
      float mnew = fmaxf(mrow[v], tmax);
      float corr = __expf(mrow[v] - mnew);
      mrow[v] = mnew;
      float rs = 0.f;
#pragma unroll
      for (int t = 0; t < 4; ++t) {
        float p = __expf(s[t][v] - mnew);
        s[t][v] = p;
        rs += p;
      }
      for (int m = 8; m; m >>= 1) rs += __shfl_xor(rs, m, 32);
      lrow[v] = lrow[v] * corr + rs;
      o0[v] *= corr;
      o1[v] *= corr;
    }

    // write P (f16) to this wave's private LDS tile
    _Float16* pbase = sP[wave];
#pragma unroll
    for (int t = 0; t < 4; ++t)
#pragma unroll
      for (int v = 0; v < 8; ++v)
        pbase[(v + half * 8) * 72 + t * 16 + l16] = (_Float16)s[t][v];

    // O += P * V
#pragma unroll
    for (int kc2 = 0; kc2 < 2; ++kc2) {
      const v8h* pp = (const v8h*)(pbase + l16 * 72 + kc2 * 32 + half * 8);
      v8h lo = pp[0], hi = pp[2];
      v16h pf;
#pragma unroll
      for (int j = 0; j < 8; ++j) { pf[j] = lo[j]; pf[j + 8] = hi[j]; }
#pragma unroll
      for (int g = 0; g < 2; ++g) {
        const v8h* vp = (const v8h*)(sVT + (g * 16 + l16) * 72 + kc2 * 32 + half * 8);
        v8h c0v = vp[0], c1v = vp[2];
        v16h vf;
#pragma unroll
        for (int j = 0; j < 8; ++j) { vf[j] = c0v[j]; vf[j + 8] = c1v[j]; }
        if (g == 0)
          o0 = __builtin_amdgcn_wmma_f32_16x16x32_f16(
              false, pf, false, vf, (short)0, o0, false, false);
        else
          o1 = __builtin_amdgcn_wmma_f32_16x16x32_f16(
              false, pf, false, vf, (short)0, o1, false, false);
      }
    }
  }

  // normalize and write out
#pragma unroll
  for (int v = 0; v < 8; ++v) {
    float inv = 1.f / lrow[v];
    int row = q0 + wave * 16 + half * 8 + v;
    float* op = O + ((size_t)(b * L + row)) * DD + h * 32;
    op[l16]      = o0[v] * inv;
    op[16 + l16] = o1[v] * inv;
  }
}

// ===========================================================================
// 7. (residual +) LayerNorm, D=256, one wave per row
// ===========================================================================
__global__ __launch_bounds__(256) void add_ln_kernel(
    const float* __restrict__ res, const float* __restrict__ y,
    const float* __restrict__ g, const float* __restrict__ b,
    float* __restrict__ out, int N)
{
  int row = blockIdx.x * 8 + (threadIdx.x >> 5);
  int lane = threadIdx.x & 31;
  if (row >= N) return;
  const float* yr = y + (size_t)row * DD;
  float vals[8];
  float s = 0.f;
#pragma unroll
  for (int j = 0; j < 8; ++j) {
    float t = yr[lane + j * 32];
    if (res) t += res[(size_t)row * DD + lane + j * 32];
    vals[j] = t;
    s += t;
  }
  for (int m = 16; m; m >>= 1) s += __shfl_xor(s, m, 32);
  float mu = s * (1.f / 256.f);
  float vs = 0.f;
#pragma unroll
  for (int j = 0; j < 8; ++j) { float d = vals[j] - mu; vs += d * d; }
  for (int m = 16; m; m >>= 1) vs += __shfl_xor(vs, m, 32);
  float rstd = rsqrtf(vs * (1.f / 256.f) + 1e-5f);
#pragma unroll
  for (int j = 0; j < 8; ++j) {
    int c = lane + j * 32;
    out[(size_t)row * DD + c] = (vals[j] - mu) * rstd * g[c] + b[c];
  }
}

// ===========================================================================
// Host-side: 3-layer post-LN transformer encoder
// ===========================================================================
static void run_encoder(const float* attn_w, const float* attn_b,
                        const float* fw1, const float* fb1,
                        const float* fw2, const float* fb2,
                        const float* ln_g, const float* ln_b,
                        const float* ng, const float* nb,
                        float* xbuf, float* q, float* k, float* v,
                        float* t, float* f, int Lcur, hipStream_t stream)
{
  const int N = BB * Lcur;
  const float scale = 0.17677669529663687f;  // 1/sqrt(32)
  dim3 blk(128);
  for (int e = 0; e < 3; ++e) {
    const float* w  = attn_w + (size_t)e * 4 * DD * DD;
    const float* wb = attn_b + (size_t)e * 4 * DD;
    gemm_kernel<<<dim3(DD / 64, N / 128), blk, 0, stream>>>(xbuf, w + 0 * (size_t)DD * DD, wb + 0 * DD, q, N, DD, DD, 0);
    gemm_kernel<<<dim3(DD / 64, N / 128), blk, 0, stream>>>(xbuf, w + 1 * (size_t)DD * DD, wb + 1 * DD, k, N, DD, DD, 0);
    gemm_kernel<<<dim3(DD / 64, N / 128), blk, 0, stream>>>(xbuf, w + 2 * (size_t)DD * DD, wb + 2 * DD, v, N, DD, DD, 0);
    attn_kernel<<<dim3(Lcur / 64, BB * HH), blk, 0, stream>>>(q, k, v, t, Lcur, scale);
    gemm_kernel<<<dim3(DD / 64, N / 128), blk, 0, stream>>>(t, w + 3 * (size_t)DD * DD, wb + 3 * DD, q, N, DD, DD, 0);
    add_ln_kernel<<<dim3(N / 8), 256, 0, stream>>>(xbuf, q, ln_g + (size_t)(e * 2) * DD, ln_b + (size_t)(e * 2) * DD, xbuf, N);
    gemm_kernel<<<dim3(FF / 64, N / 128), blk, 0, stream>>>(xbuf, fw1 + (size_t)e * DD * FF, fb1 + (size_t)e * FF, f, N, DD, FF, 1);
    gemm_kernel<<<dim3(DD / 64, N / 128), blk, 0, stream>>>(f, fw2 + (size_t)e * FF * DD, fb2 + (size_t)e * DD, t, N, FF, DD, 0);
    add_ln_kernel<<<dim3(N / 8), 256, 0, stream>>>(xbuf, t, ln_g + (size_t)(e * 2 + 1) * DD, ln_b + (size_t)(e * 2 + 1) * DD, xbuf, N);
  }
  add_ln_kernel<<<dim3(N / 8), 256, 0, stream>>>(nullptr, xbuf, ng, nb, xbuf, N);
}

// ===========================================================================
// kernel_launch
// ===========================================================================
extern "C" void kernel_launch(void* const* d_in, const int* in_sizes, int n_in,
                              void* d_out, int out_size, void* d_ws, size_t ws_size,
                              hipStream_t stream)
{
  (void)in_sizes; (void)n_in; (void)out_size; (void)ws_size;
  const float* x        = (const float*)d_in[0];
  const float* conv_w   = (const float*)d_in[1];
  const float* mask_tok = (const float*)d_in[2];
  const float* pro_w1   = (const float*)d_in[3];
  const float* pro_b1   = (const float*)d_in[4];
  const float* pro_w2   = (const float*)d_in[5];
  const float* pro_b2   = (const float*)d_in[6];
  const float* e_aw = (const float*)d_in[7];
  const float* e_ab = (const float*)d_in[8];
  const float* e_f1 = (const float*)d_in[9];
  const float* e_b1 = (const float*)d_in[10];
  const float* e_f2 = (const float*)d_in[11];
  const float* e_b2 = (const float*)d_in[12];
  const float* e_lg = (const float*)d_in[13];
  const float* e_lb = (const float*)d_in[14];
  const float* e_ng = (const float*)d_in[15];
  const float* e_nb = (const float*)d_in[16];
  const float* d_aw = (const float*)d_in[17];
  const float* d_ab = (const float*)d_in[18];
  const float* d_f1 = (const float*)d_in[19];
  const float* d_b1 = (const float*)d_in[20];
  const float* d_f2 = (const float*)d_in[21];
  const float* d_b2 = (const float*)d_in[22];
  const float* d_lg = (const float*)d_in[23];
  const float* d_lb = (const float*)d_in[24];
  const float* d_ng = (const float*)d_in[25];
  const float* d_nb = (const float*)d_in[26];

  const size_t NBIG = (size_t)BB * LL;          // 8192
  float* ws = (float*)d_ws;
  size_t p = 0;
  float* ex    = ws + p; p += NBIG * DD;
  float* tok   = ws + p; p += NBIG * DD;
  float* qb    = ws + p; p += NBIG * DD;
  float* kb    = ws + p; p += NBIG * DD;
  float* vb    = ws + p; p += NBIG * DD;
  float* tb    = ws + p; p += NBIG * DD;
  float* fb    = ws + p; p += NBIG * FF;
  float* eu    = ws + p; p += (size_t)BB * TR * DD;
  float* score = ws + p; p += NBIG;
  int* midx = (int*)(ws + p); p += (size_t)BB * TR;
  int* uidx = (int*)(ws + p); p += (size_t)BB * TR;

  embed_kernel<<<dim3(BB * LL), 256, 0, stream>>>(x, conv_w, ex);
  score_kernel<<<dim3(BB * LL / 4), 128, 0, stream>>>(ex, score);
  rank_kernel<<<dim3(BB), 1024, 0, stream>>>(score, midx, uidx);
  gather_kernel<<<dim3(BB * TR), 256, 0, stream>>>(ex, uidx, eu);
  run_encoder(e_aw, e_ab, e_f1, e_b1, e_f2, e_b2, e_lg, e_lb, e_ng, e_nb,
              eu, qb, kb, vb, tb, fb, TR, stream);
  scatter_u_kernel<<<dim3(BB * TR), 256, 0, stream>>>(eu, uidx, tok);
  scatter_m_kernel<<<dim3(BB * TR), 256, 0, stream>>>(mask_tok, midx, tok);
  run_encoder(d_aw, d_ab, d_f1, d_b1, d_f2, d_b2, d_lg, d_lb, d_ng, d_nb,
              tok, qb, kb, vb, tb, fb, LL, stream);
  gemm_kernel<<<dim3(DD / 64, (int)(NBIG / 128)), 128, 0, stream>>>(
      tok, pro_w1, pro_b1, qb, (int)NBIG, DD, DD, 2);
  gemm_kernel<<<dim3(DD / 64, (int)(NBIG / 128)), 128, 0, stream>>>(
      qb, pro_w2, pro_b2, (float*)d_out, (int)NBIG, DD, DD, 3);
}
